// MoELayer_1726576853053
// MI455X (gfx1250) — matrix operand order repaired and checked by reading, not compiled, not run on hardware
//
#include <hip/hip_runtime.h>
#include <hip/hip_bf16.h>
#include <math.h>

// ---- problem constants (from reference) ----
#define T_TOK 8192   // B*S
#define HD    1024   // hidden
#define ID    2048   // intermediate
#define NE    8      // experts

typedef __attribute__((ext_vector_type(16))) __bf16 v16bf;
typedef __attribute__((ext_vector_type(8)))  __bf16 v8bf;
typedef __attribute__((ext_vector_type(8)))  float  v8f;

static __device__ inline v16bf make_frag(v8bf lo, v8bf hi) {
  v16bf a;
#pragma unroll
  for (int j = 0; j < 8; ++j) { a[j] = lo[j]; a[j + 8] = hi[j]; }
  return a;
}

static __device__ inline float fast_sigmoid(float x) {
  return __builtin_amdgcn_rcpf(1.0f + __expf(-x));   // v_rcp_f32, no div fixup
}

// ---------------------------------------------------------------------------
// utility kernels
// ---------------------------------------------------------------------------
__global__ void moe_zero_stats(float* stats) {
  if (threadIdx.x < 18) stats[threadIdx.x] = 0.0f;
}

// 8 elements per thread: 2x b128 in, 1x b128 out
__global__ __launch_bounds__(256) void moe_cvt_bf16(const float* __restrict__ src,
                                                    __bf16* __restrict__ dst) {
  int i = blockIdx.x * 256 + threadIdx.x;
  const float4* s = (const float4*)src;
  float4 a = s[2 * i], b = s[2 * i + 1];
  v8bf o;
  o[0] = (__bf16)a.x; o[1] = (__bf16)a.y; o[2] = (__bf16)a.z; o[3] = (__bf16)a.w;
  o[4] = (__bf16)b.x; o[5] = (__bf16)b.y; o[6] = (__bf16)b.z; o[7] = (__bf16)b.w;
  *(v8bf*)(dst + 8 * i) = o;
}

// ---------------------------------------------------------------------------
// router: logits = x @ gate_w^T, softmax, top-2, combine weights + aux stats
// one token per wave (wave32), 8 waves per block
// stats[0..7]=top2 counts, [8..15]=prob sums, [16]=sum lse^2, [17]=sum entropy
// ---------------------------------------------------------------------------
__global__ __launch_bounds__(256) void moe_router(const float* __restrict__ x,
                                                  const float* __restrict__ gw,
                                                  float* __restrict__ combine,
                                                  float* __restrict__ stats) {
  const int wave = threadIdx.x >> 5;
  const int lane = threadIdx.x & 31;
  const int t = blockIdx.x * 8 + wave;
  const float* xr = x + (size_t)t * HD;

  float acc[NE] = {};
  for (int j = lane; j < HD; j += 32) {
    float xv = xr[j];
#pragma unroll
    for (int e = 0; e < NE; ++e) acc[e] += xv * gw[e * HD + j];
  }
#pragma unroll
  for (int e = 0; e < NE; ++e) {
#pragma unroll
    for (int off = 16; off > 0; off >>= 1) acc[e] += __shfl_xor(acc[e], off);
  }
  if (lane == 0) {
    float mx = acc[0];
#pragma unroll
    for (int e = 1; e < NE; ++e) mx = fmaxf(mx, acc[e]);
    float p[NE]; float s = 0.0f;
#pragma unroll
    for (int e = 0; e < NE; ++e) { p[e] = __expf(acc[e] - mx); s += p[e]; }
    float lse = mx + __logf(s);
    float inv = __builtin_amdgcn_rcpf(s);
    float ent = 0.0f;
#pragma unroll
    for (int e = 0; e < NE; ++e) { p[e] *= inv; ent -= p[e] * __logf(p[e] + 1e-9f); }
    int i1 = 0;
#pragma unroll
    for (int e = 1; e < NE; ++e) if (p[e] > p[i1]) i1 = e;
    int i2 = (i1 == 0) ? 1 : 0;
#pragma unroll
    for (int e = 0; e < NE; ++e) if (e != i1 && p[e] > p[i2]) i2 = e;
    float rden = __builtin_amdgcn_rcpf(p[i1] + p[i2] + 1e-9f);
#pragma unroll
    for (int e = 0; e < NE; ++e) combine[t * NE + e] = 0.0f;
    combine[t * NE + i1] = p[i1] * rden;
    combine[t * NE + i2] = p[i2] * rden;
    atomicAdd(&stats[i1], 1.0f);
    atomicAdd(&stats[i2], 1.0f);
#pragma unroll
    for (int e = 0; e < NE; ++e) atomicAdd(&stats[8 + e], p[e]);
    atomicAdd(&stats[16], lse * lse);
    atomicAdd(&stats[17], ent);
  }
}

// ---------------------------------------------------------------------------
// fused gate+up GEMM:  hact[t,i] = silu(x.Wg[i,:]) * (x.Wu[i,:])  (bf16 out)
// WG = 256 threads = 8 waves; tile 256(M) x 64(N); each wave: 32(M) x 64(N)
// => every B fragment feeds 2 WMMAs. Double-buffered LDS (1 barrier / K-step),
// weights prefetched f32->regs->bf16 LDS one step ahead.
// ---------------------------------------------------------------------------
__global__ __launch_bounds__(256) void moe_gateup(const __bf16* __restrict__ xb,
                                                  const float* __restrict__ wg,
                                                  const float* __restrict__ wu,
                                                  __bf16* __restrict__ hact) {
  __shared__ __align__(16) __bf16 ldsG[2 * 64 * 40];
  __shared__ __align__(16) __bf16 ldsU[2 * 64 * 40];
  const int tid  = threadIdx.x;
  const int wave = tid >> 5;
  const int lane = tid & 31;
  const int g    = lane >> 4;       // half-wave group
  const int lr   = lane & 15;
  const int t0   = blockIdx.x * 256;
  const int n0   = blockIdx.y * 64;
  const int lrow   = tid >> 2;       // 0..63 tile row for cooperative load
  const int lchunk = (tid & 3) * 8;  // 0/8/16/24
  const int lsoff  = lrow * 40 + lchunk;

  v8f accG[2][4] = {}; v8f accU[2][4] = {};
  const __bf16* arow0 = xb + (size_t)(t0 + wave * 32 + lr) * HD;
  const __bf16* arow1 = arow0 + (size_t)16 * HD;
  const float* gptr = wg + (size_t)(n0 + lrow) * HD + lchunk;
  const float* uptr = wu + (size_t)(n0 + lrow) * HD + lchunk;

  float pg[8], pu[8];
  // chunk 0 -> buf0, then prefetch chunk 1
#pragma unroll
  for (int j = 0; j < 8; ++j) { pg[j] = gptr[j]; pu[j] = uptr[j]; }
#pragma unroll
  for (int j = 0; j < 8; ++j) { ldsG[lsoff + j] = (__bf16)pg[j]; ldsU[lsoff + j] = (__bf16)pu[j]; }
#pragma unroll
  for (int j = 0; j < 8; ++j) { pg[j] = gptr[32 + j]; pu[j] = uptr[32 + j]; }
  __syncthreads();

  for (int k0 = 0; k0 < HD; k0 += 32) {
    const int cur = (k0 >> 5) & 1;
    const int buf = cur * 2560;

    v16bf a0 = make_frag(*(const v8bf*)(arow0 + k0 + g * 8),
                         *(const v8bf*)(arow0 + k0 + 16 + g * 8));
    v16bf a1 = make_frag(*(const v8bf*)(arow1 + k0 + g * 8),
                         *(const v8bf*)(arow1 + k0 + 16 + g * 8));
#pragma unroll
    for (int ns = 0; ns < 4; ++ns) {
      const __bf16* bgrow = &ldsG[buf + (ns * 16 + lr) * 40 + g * 8];
      const __bf16* burow = &ldsU[buf + (ns * 16 + lr) * 40 + g * 8];
      v16bf bg = make_frag(*(const v8bf*)bgrow, *(const v8bf*)(bgrow + 16));
      v16bf bu = make_frag(*(const v8bf*)burow, *(const v8bf*)(burow + 16));
      accG[0][ns] = __builtin_amdgcn_wmma_f32_16x16x32_bf16(false, a0, false, bg, (short)0,
                                                            accG[0][ns], false, false);
      accG[1][ns] = __builtin_amdgcn_wmma_f32_16x16x32_bf16(false, a1, false, bg, (short)0,
                                                            accG[1][ns], false, false);
      accU[0][ns] = __builtin_amdgcn_wmma_f32_16x16x32_bf16(false, a0, false, bu, (short)0,
                                                            accU[0][ns], false, false);
      accU[1][ns] = __builtin_amdgcn_wmma_f32_16x16x32_bf16(false, a1, false, bu, (short)0,
                                                            accU[1][ns], false, false);
    }

    if (k0 + 32 < HD) {                     // stage chunk k0+32 into other buffer
      const int obuf = (cur ^ 1) * 2560;
#pragma unroll
      for (int j = 0; j < 8; ++j) {
        ldsG[obuf + lsoff + j] = (__bf16)pg[j];
        ldsU[obuf + lsoff + j] = (__bf16)pu[j];
      }
      if (k0 + 64 < HD) {                   // prefetch chunk k0+64 (lands next iter)
#pragma unroll
        for (int j = 0; j < 8; ++j) { pg[j] = gptr[k0 + 64 + j]; pu[j] = uptr[k0 + 64 + j]; }
      }
    }
    __syncthreads();
  }

#pragma unroll
  for (int ms = 0; ms < 2; ++ms) {
#pragma unroll
    for (int ns = 0; ns < 4; ++ns) {
#pragma unroll
      for (int r = 0; r < 8; ++r) {
        float gv = accG[ms][ns][r];
        float h  = gv * fast_sigmoid(gv) * accU[ms][ns][r];   // silu(g) * u
        int row = t0 + wave * 32 + ms * 16 + g * 8 + r;
        int col = n0 + ns * 16 + lr;
        hact[(size_t)row * ID + col] = (__bf16)h;
      }
    }
  }
}

// ---------------------------------------------------------------------------
// down-proj GEMM + combine scaling: out[t,h] (+)= scale_t * (hact . Wd[h,:])
// accumulate==0: scale = sigmoid(*sgate) and plain store (shared expert first)
// accumulate==1: scale = combine[t, expert] and read-modify-write
// ---------------------------------------------------------------------------
__global__ __launch_bounds__(256) void moe_down(const __bf16* __restrict__ hact,
                                                const float* __restrict__ wd,
                                                const float* __restrict__ combine,
                                                const float* __restrict__ sgate,
                                                float* __restrict__ out,
                                                int expert, int accumulate) {
  __shared__ __align__(16) __bf16 ldsB[2 * 64 * 40];
  const int tid  = threadIdx.x;
  const int wave = tid >> 5;
  const int lane = tid & 31;
  const int g    = lane >> 4;
  const int lr   = lane & 15;
  const int t0   = blockIdx.x * 256;
  const int n0   = blockIdx.y * 64;
  const int lrow   = tid >> 2;
  const int lchunk = (tid & 3) * 8;
  const int lsoff  = lrow * 40 + lchunk;

  v8f acc[2][4] = {};
  const __bf16* arow0 = hact + (size_t)(t0 + wave * 32 + lr) * ID;
  const __bf16* arow1 = arow0 + (size_t)16 * ID;
  const float* wptr = wd + (size_t)(n0 + lrow) * ID + lchunk;

  float pw[8];
#pragma unroll
  for (int j = 0; j < 8; ++j) pw[j] = wptr[j];
#pragma unroll
  for (int j = 0; j < 8; ++j) ldsB[lsoff + j] = (__bf16)pw[j];
#pragma unroll
  for (int j = 0; j < 8; ++j) pw[j] = wptr[32 + j];
  __syncthreads();

  for (int k0 = 0; k0 < ID; k0 += 32) {
    const int cur = (k0 >> 5) & 1;
    const int buf = cur * 2560;

    v16bf a0 = make_frag(*(const v8bf*)(arow0 + k0 + g * 8),
                         *(const v8bf*)(arow0 + k0 + 16 + g * 8));
    v16bf a1 = make_frag(*(const v8bf*)(arow1 + k0 + g * 8),
                         *(const v8bf*)(arow1 + k0 + 16 + g * 8));
#pragma unroll
    for (int ns = 0; ns < 4; ++ns) {
      const __bf16* brow = &ldsB[buf + (ns * 16 + lr) * 40 + g * 8];
      v16bf b = make_frag(*(const v8bf*)brow, *(const v8bf*)(brow + 16));
      acc[0][ns] = __builtin_amdgcn_wmma_f32_16x16x32_bf16(false, a0, false, b, (short)0,
                                                           acc[0][ns], false, false);
      acc[1][ns] = __builtin_amdgcn_wmma_f32_16x16x32_bf16(false, a1, false, b, (short)0,
                                                           acc[1][ns], false, false);
    }

    if (k0 + 32 < ID) {
      const int obuf = (cur ^ 1) * 2560;
#pragma unroll
      for (int j = 0; j < 8; ++j) ldsB[obuf + lsoff + j] = (__bf16)pw[j];
      if (k0 + 64 < ID) {
#pragma unroll
        for (int j = 0; j < 8; ++j) pw[j] = wptr[k0 + 64 + j];
      }
    }
    __syncthreads();
  }

  float sigm = fast_sigmoid(sgate[0]);
#pragma unroll
  for (int ms = 0; ms < 2; ++ms) {
    float sc[8];
#pragma unroll
    for (int r = 0; r < 8; ++r) {
      int row = t0 + wave * 32 + ms * 16 + g * 8 + r;
      sc[r] = accumulate ? combine[row * NE + expert] : sigm;
    }
#pragma unroll
    for (int ns = 0; ns < 4; ++ns) {
#pragma unroll
      for (int r = 0; r < 8; ++r) {
        int row = t0 + wave * 32 + ms * 16 + g * 8 + r;
        int col = n0 + ns * 16 + lr;
        size_t idx = (size_t)row * HD + col;
        float v = sc[r] * acc[ms][ns][r];
        if (accumulate) out[idx] += v; else out[idx] = v;
      }
    }
  }
}

// ---------------------------------------------------------------------------
// aux-loss finalize
// ---------------------------------------------------------------------------
__global__ void moe_finalize(const float* __restrict__ stats, float* __restrict__ out) {
  if (threadIdx.x == 0) {
    float lb = 0.0f, usage = 0.0f;
#pragma unroll
    for (int e = 0; e < NE; ++e) {
      float tpe = stats[e] / ((float)(T_TOK * 2) + 1e-9f);
      float avg = stats[8 + e] / (float)T_TOK;
      lb += tpe * avg;
      usage += (tpe > 0.01f) ? 1.0f : 0.0f;
    }
    lb *= (float)NE;
    float z    = (stats[16] / (float)T_TOK) * 0.001f;
    float ent  = stats[17] / (float)T_TOK;
    float entl = (__logf((float)NE) - ent) * 0.01f;
    float util = (1.0f - usage / (float)NE) * 0.1f;
    out[(size_t)T_TOK * HD] = lb + z + entl + util;
  }
}

// ---------------------------------------------------------------------------
extern "C" void kernel_launch(void* const* d_in, const int* in_sizes, int n_in,
                              void* d_out, int out_size, void* d_ws, size_t ws_size,
                              hipStream_t stream) {
  const float* x     = (const float*)d_in[0];  // [T,H]
  const float* gatew = (const float*)d_in[1];  // [E,H]
  const float* eg    = (const float*)d_in[2];  // [E,I,H]
  const float* eu    = (const float*)d_in[3];  // [E,I,H]
  const float* ed    = (const float*)d_in[4];  // [E,H,I]
  const float* sgw   = (const float*)d_in[5];  // [I,H]
  const float* suw   = (const float*)d_in[6];  // [I,H]
  const float* sdw   = (const float*)d_in[7];  // [H,I]
  const float* sscal = (const float*)d_in[8];  // [1]
  float* out = (float*)d_out;                  // [T*H] final + [1] aux

  // workspace layout (≈50.6 MB)
  char* w = (char*)d_ws;
  __bf16* x_bf    = (__bf16*)w;                                   // T*H bf16 = 16 MB
  __bf16* hact    = (__bf16*)(w + (size_t)16777216);              // T*I bf16 = 32 MB
  float*  combine = (float*)(w + (size_t)16777216 + 33554432);    // T*E f32
  float*  stats   = combine + (size_t)T_TOK * NE;                 // 18 f32

  dim3 blk(256);
  moe_zero_stats<<<1, 32, 0, stream>>>(stats);
  moe_cvt_bf16<<<(T_TOK * HD) / (256 * 8), blk, 0, stream>>>(x, x_bf);
  moe_router<<<T_TOK / 8, blk, 0, stream>>>(x, gatew, combine, stats);

  dim3 gGU(T_TOK / 256, ID / 64);   // (32, 32)
  dim3 gDN(T_TOK / 256, HD / 64);   // (32, 16)

  // shared expert first: stores (overwrites poisoned d_out) with sigmoid gate scale
  moe_gateup<<<gGU, blk, 0, stream>>>(x_bf, sgw, suw, hact);
  moe_down<<<gDN, blk, 0, stream>>>(hact, sdw, combine, sscal, out, 0, 0);

  // routed experts: accumulate with per-token combine weights
  for (int e = 0; e < NE; ++e) {
    moe_gateup<<<gGU, blk, 0, stream>>>(x_bf, eg + (size_t)e * ID * HD,
                                        eu + (size_t)e * ID * HD, hact);
    moe_down<<<gDN, blk, 0, stream>>>(hact, ed + (size_t)e * HD * ID,
                                      combine, sscal, out, e, 1);
  }

  moe_finalize<<<1, 32, 0, stream>>>(stats, out);
}
